// SemanticVQCompressor_26439818674911
// MI455X (gfx1250) — matrix operand order repaired and checked by reading, not compiled, not run on hardware
//
#include <hip/hip_runtime.h>

// ---------------- problem constants (match reference) ----------------
#define HIDDEN   4096
#define CODE_DIM 256
#define KCODES   4096
#define NTOK     16384          // B*S = 4*4096
#define BETA     0.25f

// ---------------- CDNA5 WMMA types ----------------
typedef __attribute__((ext_vector_type(16))) __bf16 bf16x16;
typedef __attribute__((ext_vector_type(8)))  float  f32x8;

union BFrag  { bf16x16 v; uint4 q[2]; unsigned short s[16]; };

__device__ __forceinline__ unsigned short f2bf(float x) {
    unsigned int u = __float_as_uint(x);
    unsigned int r = u + 0x7FFFu + ((u >> 16) & 1u);   // round-to-nearest-even
    return (unsigned short)(r >> 16);
}

// ---- CDNA5 async global->LDS copy (ASYNCcnt) -------------------------------
// VDST = VGPR with LDS byte address (low 32 bits of a generic LDS pointer),
// VADDR = 64-bit global address. 16B granularity, both sides 16B aligned.
__device__ __forceinline__ void async_copy16(unsigned ldsByteOff, const void* gptr) {
    asm volatile("global_load_async_to_lds_b128 %0, %1, off"
                 :: "v"(ldsByteOff), "v"((unsigned long long)(uintptr_t)gptr)
                 : "memory");
}
__device__ __forceinline__ void wait_async0() {
    asm volatile("s_wait_asynccnt 0x0" ::: "memory");
}

// A-matrix 16x32 bf16 fragment from an f32 row (row-major, this lane's row).
// Contiguous k order used here (k = 16*half + s), same as B: lanes 0-15 hold K 0..15; lanes 16-31 hold K 16..31.
__device__ __forceinline__ bf16x16 load_a_frag_f32(const float* __restrict__ rowPtr,
                                                   int k0, int half) {
    const int kc0 = k0 + 16 * half;
    const int kc1 = k0 + 16 * half + 8;
    float4 a0 = *(const float4*)(rowPtr + kc0);
    float4 a1 = *(const float4*)(rowPtr + kc0 + 4);
    float4 b0 = *(const float4*)(rowPtr + kc1);
    float4 b1 = *(const float4*)(rowPtr + kc1 + 4);
    BFrag f;
    f.s[0] = f2bf(a0.x); f.s[1] = f2bf(a0.y); f.s[2]  = f2bf(a0.z); f.s[3]  = f2bf(a0.w);
    f.s[4] = f2bf(a1.x); f.s[5] = f2bf(a1.y); f.s[6]  = f2bf(a1.z); f.s[7]  = f2bf(a1.w);
    f.s[8] = f2bf(b0.x); f.s[9] = f2bf(b0.y); f.s[10] = f2bf(b0.z); f.s[11] = f2bf(b0.w);
    f.s[12]= f2bf(b1.x); f.s[13]= f2bf(b1.y); f.s[14] = f2bf(b1.z); f.s[15] = f2bf(b1.w);
    return f.v;
}

// B-matrix 32x16 bf16 fragment from LDS (16 contiguous bf16, 16B aligned).
// Contiguous k order used here, same as A: lanes 0-15 hold K 0..15, lanes 16-31 hold K 16..31; lane%16 = column.
__device__ __forceinline__ bf16x16 load_b_frag_lds(const unsigned short* p) {
    BFrag r;
    r.q[0] = ((const uint4*)p)[0];
    r.q[1] = ((const uint4*)p)[1];
    return r.v;
}

// =====================================================================
// Fused bf16-WMMA GEMM:  Out[m, nColBase+n] = A[row(m), :] * B[n, :]^T + bias[n]
//   A: f32 [*, KTOT] row-major; row(m) = GATHER ? gidx[m] : m
//   B: bf16(u16) [Ncols_total, KTOT] row-major (pre-converted, L2-resident)
// Block: 256 thr (8 waves). Tile: 128 rows x 256 cols.
// B k-slabs (32x256 bf16) double-buffered in LDS via async global->LDS copies.
// =====================================================================
template<int KTOT, bool GATHER>
__global__ __launch_bounds__(256)
void gemm_bf16_wmma(const float* __restrict__ Asrc,
                    const unsigned short* __restrict__ Bsrc,
                    const float* __restrict__ bias,
                    const int* __restrict__ gidx,
                    float* __restrict__ Out, int ldo) {
    constexpr int BSTR = 40;                       // 32 k + 8 pad (80B stride)
    constexpr int KB   = KTOT / 32;
    __shared__ unsigned short Bl[2][256 * BSTR];   // 40 KB, double buffered

    const int t    = threadIdx.x;
    const int wave = t >> 5;
    const int lane = t & 31;
    const int half = lane >> 4;
    const int lc   = lane & 15;
    const int mBase    = blockIdx.x * 128 + wave * 16;
    const int nColBase = blockIdx.y * 256;

    const int arow = GATHER ? gidx[mBase + lc] : (mBase + lc);
    const float* aRowPtr = Asrc + (size_t)arow * KTOT;

    // thread t stages output-column t of the k-slab: 32 bf16 = 64B = 4x B128
    const char* gColBase = (const char*)(Bsrc + (size_t)(nColBase + t) * KTOT);
    auto stage = [&](int buf, int kb) {
        const char* g = gColBase + (size_t)kb * 64;          // 32 bf16 per k-slab
        unsigned l = (unsigned)(uintptr_t)&Bl[buf][t * BSTR];
#pragma unroll
        for (int j = 0; j < 4; ++j) async_copy16(l + j * 16, g + j * 16);
    };

    f32x8 acc[16];
#pragma unroll
    for (int nt = 0; nt < 16; ++nt) {
        float bv = bias[nColBase + nt * 16 + lc];
#pragma unroll
        for (int v = 0; v < 8; ++v) acc[nt][v] = bv;
    }

    stage(0, 0);
    wait_async0();
    __syncthreads();

    for (int kb = 0; kb < KB; ++kb) {
        if (kb + 1 < KB) stage((kb + 1) & 1, kb + 1);        // prefetch next slab
        const unsigned short* cur = &Bl[kb & 1][0];
        bf16x16 afrag = load_a_frag_f32(aRowPtr, kb * 32, half);
#pragma unroll
        for (int g = 0; g < 2; ++g) {                        // batch 8 B-frags, then 8 WMMAs
            bf16x16 bfr[8];
#pragma unroll
            for (int j = 0; j < 8; ++j)
                bfr[j] = load_b_frag_lds(&cur[((g * 8 + j) * 16 + lc) * BSTR + 16 * half]);
#pragma unroll
            for (int j = 0; j < 8; ++j)
                acc[g * 8 + j] = __builtin_amdgcn_wmma_f32_16x16x32_bf16(
                    false, afrag, false, bfr[j], (short)0, acc[g * 8 + j], false, false);
        }
        wait_async0();                                       // prefetch overlapped compute
        __syncthreads();
    }

#pragma unroll
    for (int nt = 0; nt < 16; ++nt)
#pragma unroll
        for (int v = 0; v < 8; ++v)
            Out[(size_t)(mBase + v + 8 * half) * ldo + nColBase + nt * 16 + lc] = acc[nt][v];
}

// =====================================================================
// Nearest-codebook search with WMMA: dist = e2[k] - 2*(z . e_k)  (x2 constant/row)
// Block: 256 thr, 128 token rows. z A-fragments live in VGPRs for whole kernel.
// Codebook (pre-converted bf16) staged in 32-code slabs, double-buffered async.
// =====================================================================
__global__ __launch_bounds__(256)
void vq_argmin_wmma(const float* __restrict__ Z,
                    const unsigned short* __restrict__ cb16,
                    const float* __restrict__ e2g,
                    int* __restrict__ idxOut) {
    constexpr int CSTR  = 264;                     // 256 + 8 pad (528B stride)
    constexpr int SLABS = KCODES / 32;
    __shared__ unsigned short Cl[2][32 * CSTR];    // ~33 KB, double buffered

    const int t    = threadIdx.x;
    const int wave = t >> 5;
    const int lane = t & 31;
    const int half = lane >> 4;
    const int lc   = lane & 15;
    const int mBase = blockIdx.x * 128 + wave * 16;

    const float* zRow = Z + (size_t)(mBase + lc) * CODE_DIM;
    bf16x16 afr[8];
#pragma unroll
    for (int kb = 0; kb < 8; ++kb) afr[kb] = load_a_frag_f32(zRow, kb * 32, half);

    // thread t stages (code t>>3, 64B segment t&7) of the 32-code slab
    const int sc  = t >> 3, ss = t & 7;
    auto stage = [&](int buf, int slab) {
        const char* g = (const char*)(cb16 + ((size_t)(slab * 32 + sc)) * CODE_DIM) + ss * 64;
        unsigned l = (unsigned)(uintptr_t)&Cl[buf][sc * CSTR + ss * 32];
#pragma unroll
        for (int j = 0; j < 4; ++j) async_copy16(l + j * 16, g + j * 16);
    };

    float minv[8]; int mini[8];
#pragma unroll
    for (int v = 0; v < 8; ++v) { minv[v] = 3.4e38f; mini[v] = 0; }

    stage(0, 0);
    wait_async0();
    __syncthreads();

    for (int slab = 0; slab < SLABS; ++slab) {
        if (slab + 1 < SLABS) stage((slab + 1) & 1, slab + 1);
        const unsigned short* cur = &Cl[slab & 1][0];
        const int cBase = slab * 32;
#pragma unroll
        for (int nt = 0; nt < 2; ++nt) {
            const int codeN = cBase + nt * 16 + lc;          // this lane's column (code)
            const float e2v = e2g[codeN];
            bf16x16 bfr[8];
#pragma unroll
            for (int kb = 0; kb < 8; ++kb)
                bfr[kb] = load_b_frag_lds(&cur[(nt * 16 + lc) * CSTR + kb * 32 + 16 * half]);
            f32x8 acc;
#pragma unroll
            for (int v = 0; v < 8; ++v) acc[v] = 0.0f;
#pragma unroll
            for (int kb = 0; kb < 8; ++kb)
                acc = __builtin_amdgcn_wmma_f32_16x16x32_bf16(
                    false, afr[kb], false, bfr[kb], (short)0, acc, false, false);
#pragma unroll
            for (int v = 0; v < 8; ++v) {
                float d = e2v - 2.0f * acc[v];
                if (d < minv[v]) { minv[v] = d; mini[v] = codeN; }
            }
        }
        wait_async0();
        __syncthreads();
    }

    // reduce (min, argmin) across the 16 lanes of each half (same rows, disjoint codes)
#pragma unroll
    for (int m = 1; m < 16; m <<= 1) {
#pragma unroll
        for (int v = 0; v < 8; ++v) {
            float ov = __shfl_xor(minv[v], m, 32);
            int   oi = __shfl_xor(mini[v], m, 32);
            if (ov < minv[v] || (ov == minv[v] && oi < mini[v])) { minv[v] = ov; mini[v] = oi; }
        }
    }
    if (lc == 0) {
#pragma unroll
        for (int v = 0; v < 8; ++v) idxOut[mBase + 8 * half + v] = mini[v];
    }
}

// ---------------- small helper kernels ----------------
__global__ void cvt_f32_bf16(const float* __restrict__ src, unsigned short* __restrict__ dst, int n) {
    int i = (blockIdx.x * blockDim.x + threadIdx.x) * 4;
    if (i < n) {
        float4 f = *(const float4*)(src + i);
        dst[i + 0] = f2bf(f.x); dst[i + 1] = f2bf(f.y);
        dst[i + 2] = f2bf(f.z); dst[i + 3] = f2bf(f.w);
    }
}

__global__ void e2_kernel(const float* __restrict__ cb, float* __restrict__ e2g) {
    int c = blockIdx.x * 256 + threadIdx.x;
    const float* r = cb + (size_t)c * CODE_DIM;
    float s = 0.f;
    for (int i = 0; i < CODE_DIM; i += 4) {
        float4 f = *(const float4*)(r + i);
        s += f.x * f.x + f.y * f.y + f.z * f.z + f.w * f.w;
    }
    e2g[c] = s;
}

// scal[0]=logZ, scal[1]=vq_sum (zeroed), scal[2]=prior_sum (zeroed)
__global__ void logz_zero_kernel(const float* __restrict__ logits, float* __restrict__ scal) {
    __shared__ float red[256];
    int t = threadIdx.x;
    float mx = -3.4e38f;
    for (int i = t; i < KCODES; i += 256) mx = fmaxf(mx, logits[i]);
    red[t] = mx; __syncthreads();
    for (int s = 128; s > 0; s >>= 1) { if (t < s) red[t] = fmaxf(red[t], red[t + s]); __syncthreads(); }
    mx = red[0]; __syncthreads();
    float se = 0.f;
    for (int i = t; i < KCODES; i += 256) se += __expf(logits[i] - mx);
    red[t] = se; __syncthreads();
    for (int s = 128; s > 0; s >>= 1) { if (t < s) red[t] += red[t + s]; __syncthreads(); }
    if (t == 0) { scal[0] = mx + __logf(red[0]); scal[1] = 0.f; scal[2] = 0.f; }
}

// per-token: sum (z - codebook[idx])^2, sum prior[idx]; also export idx to d_out tail
__global__ __launch_bounds__(256)
void vq_loss_kernel(const float* __restrict__ Z, const float* __restrict__ cb,
                    const int* __restrict__ idx, const float* __restrict__ logits,
                    float* __restrict__ scal, int* __restrict__ idxOutTail) {
    __shared__ float r1[256], r2[256];
    int tok = blockIdx.x * 256 + threadIdx.x;
    int c = idx[tok];
    idxOutTail[tok] = c;
    const float* z = Z  + (size_t)tok * CODE_DIM;
    const float* e = cb + (size_t)c   * CODE_DIM;
    float s = 0.f;
    for (int i = 0; i < CODE_DIM; i += 4) {
        float4 a = *(const float4*)(z + i);
        float4 b = *(const float4*)(e + i);
        float dx = a.x - b.x, dy = a.y - b.y, dz = a.z - b.z, dw = a.w - b.w;
        s += dx * dx + dy * dy + dz * dz + dw * dw;
    }
    r1[threadIdx.x] = s;
    r2[threadIdx.x] = logits[c];
    __syncthreads();
    for (int st = 128; st > 0; st >>= 1) {
        if (threadIdx.x < st) { r1[threadIdx.x] += r1[threadIdx.x + st];
                                r2[threadIdx.x] += r2[threadIdx.x + st]; }
        __syncthreads();
    }
    if (threadIdx.x == 0) { atomicAdd(&scal[1], r1[0]); atomicAdd(&scal[2], r2[0]); }
}

__global__ void finalize_kernel(const float* __restrict__ scal,
                                float* __restrict__ outBits, float* __restrict__ outVq) {
    float logZ = scal[0], vq = scal[1], pr = scal[2];
    *outBits = ((float)NTOK * logZ - pr) * 1.4426950408889634f;       // / ln 2
    *outVq   = vq * (1.0f + BETA) / (float)((size_t)NTOK * CODE_DIM); // 1.25 * mean
}

// =====================================================================
extern "C" void kernel_launch(void* const* d_in, const int* in_sizes, int n_in,
                              void* d_out, int out_size, void* d_ws, size_t ws_size,
                              hipStream_t stream) {
    (void)in_sizes; (void)n_in; (void)out_size; (void)ws_size;
    const float* embed    = (const float*)d_in[0];   // [4,4096,4096]
    const float* W_pre    = (const float*)d_in[1];   // [256,4096]
    const float* b_pre    = (const float*)d_in[2];   // [256]
    const float* codebook = (const float*)d_in[3];   // [4096,256]
    const float* W_post   = (const float*)d_in[4];   // [4096,256]
    const float* b_post   = (const float*)d_in[5];   // [4096]
    const float* prior    = (const float*)d_in[6];   // [4096]
    float* out = (float*)d_out;                      // [embed_hat | bits | vq | idx]
    char*  ws  = (char*)d_ws;

    float*          zbuf    = (float*)         (ws);                         // 16 MB
    unsigned short* wpre16  = (unsigned short*)(ws + (16u << 20));           //  2 MB
    unsigned short* wpost16 = (unsigned short*)(ws + (18u << 20));           //  2 MB
    unsigned short* cb16    = (unsigned short*)(ws + (20u << 20));           //  2 MB
    float*          e2buf   = (float*)         (ws + (22u << 20));           // 16 KB
    int*            idxbuf  = (int*)           (ws + (22u << 20) + 16384u);  // 64 KB
    float*          scal    = (float*)         (ws + (22u << 20) + 81920u);  // 3 floats

    const size_t HAT = (size_t)NTOK * HIDDEN;        // 67108864

    // prep: bf16 weight/codebook copies, ||e||^2, log-sum-exp + zero accumulators
    cvt_f32_bf16<<<(CODE_DIM * HIDDEN) / 1024, 256, 0, stream>>>(W_pre,    wpre16,  CODE_DIM * HIDDEN);
    cvt_f32_bf16<<<(HIDDEN * CODE_DIM) / 1024, 256, 0, stream>>>(W_post,   wpost16, HIDDEN * CODE_DIM);
    cvt_f32_bf16<<<(KCODES * CODE_DIM) / 1024, 256, 0, stream>>>(codebook, cb16,    KCODES * CODE_DIM);
    e2_kernel<<<KCODES / 256, 256, 0, stream>>>(codebook, e2buf);
    logz_zero_kernel<<<1, 256, 0, stream>>>(prior, scal);

    // z = embed @ W_pre^T + b_pre     (M=16384, N=256, K=4096)
    gemm_bf16_wmma<HIDDEN, false><<<dim3(NTOK / 128, 1), 256, 0, stream>>>(
        embed, wpre16, b_pre, nullptr, zbuf, CODE_DIM);

    // idx = argmin_k ||z - e_k||^2
    vq_argmin_wmma<<<NTOK / 128, 256, 0, stream>>>(zbuf, cb16, e2buf, idxbuf);

    // vq/prior partial sums + idx export
    vq_loss_kernel<<<NTOK / 256, 256, 0, stream>>>(zbuf, codebook, idxbuf, prior,
                                                   scal, (int*)(out + HAT + 2));

    // embed_hat = codebook[idx] @ W_post^T + b_post   (M=16384, N=4096, K=256)
    gemm_bf16_wmma<CODE_DIM, true><<<dim3(NTOK / 128, HIDDEN / 256), 256, 0, stream>>>(
        codebook, wpost16, b_post, idxbuf, out, HIDDEN);

    finalize_kernel<<<1, 1, 0, stream>>>(scal, out + HAT, out + HAT + 1);
}
